// Net_58712202936754
// MI455X (gfx1250) — compile-verified, matrix operand order checked by reading, not run on hardware
//
#include <hip/hip_runtime.h>

// ---------------------------------------------------------------------------
// CDNA5 (gfx1250) WMMA helpers: f16 inputs, f32 accumulate, 16x16x32 tiles.
// ---------------------------------------------------------------------------
typedef __attribute__((ext_vector_type(16))) _Float16 v16h;
typedef __attribute__((ext_vector_type(8)))  float    v8f;

union Frag { v16h v; unsigned u[8]; };

__device__ __forceinline__ unsigned pack2(float a, float b) {
  union { _Float16 h[2]; unsigned u; } p;
  p.h[0] = (_Float16)a; p.h[1] = (_Float16)b;
  return p.u;
}

// K-offset of VGPR pair q (0..7) for the 16-bit A/B fragment layout.
// lanes 0-15: K = {0..7, 16..23}; lanes 16-31: +8.
__device__ __forceinline__ int kqoff(int q, int hi) {
  return ((q < 4) ? (2 * q) : (16 + 2 * (q - 4))) + 8 * hi;
}

__device__ __forceinline__ v8f wmma_f16(const Frag& a, const Frag& b, v8f c) {
  return __builtin_amdgcn_wmma_f32_16x16x32_f16(false, a.v, false, b.v,
                                                (short)0, c, false, false);
}

// ---------------------------------------------------------------------------
// Tensor Data Mover: async DMA of a 2D f16 tile [rows x tileK] (row stride
// strideK halves) from global into LDS.  Built per CDNA5 D# layout
// (group0: count/lds/global addr; group1: dims/strides).  Guarded: falls back
// to a cooperative copy when the builtin is unavailable.
// ---------------------------------------------------------------------------
typedef __attribute__((ext_vector_type(4))) unsigned tdm_u4;
typedef __attribute__((ext_vector_type(8))) int      tdm_i8;
typedef __attribute__((ext_vector_type(4))) int      tdm_i4;

#if defined(__has_builtin)
#if __has_builtin(__builtin_amdgcn_tensor_load_to_lds)
#define HAVE_TDM 1
#endif
#endif

#ifdef HAVE_TDM
__device__ __forceinline__ void tdm_load_a_tile(const _Float16* gsrc,
                                                unsigned ldsOff, int tileK,
                                                int rows, int strideK) {
  const size_t ga = (size_t)gsrc;
  tdm_u4 g0;
  g0[0] = 1u;                                            // count=1 (valid user D#)
  g0[1] = ldsOff;                                        // lds_addr (bytes)
  g0[2] = (unsigned)(ga & 0xFFFFFFFFu);                  // global_addr[31:0]
  g0[3] = (unsigned)((ga >> 32) & 0x01FFFFFFu) | (2u << 30);  // addr[56:32]|type=2
  tdm_i8 g1;
  g1[0] = 0x00010000;                                    // data_size = 2 bytes
  g1[1] = (int)((strideK & 0xFFFF) << 16);               // tensor_dim0[15:0]
  g1[2] = (int)((((unsigned)strideK >> 16) & 0xFFFFu) |  // tensor_dim0[31:16]
                (((unsigned)rows & 0xFFFFu) << 16));     // tensor_dim1[15:0]
  g1[3] = (int)(((unsigned)tileK & 0xFFFFu) << 16);      // tile_dim0
  g1[4] = (int)((unsigned)rows & 0xFFFFu);               // tile_dim1 (tile_dim2=0)
  g1[5] = strideK;                                       // tensor_dim0_stride[31:0]
  g1[6] = 0;
  g1[7] = 0;
  tdm_i4 z4 = {0, 0, 0, 0};
#if defined(__clang_major__) && (__clang_major__ >= 23)
  tdm_i8 z8 = {0, 0, 0, 0, 0, 0, 0, 0};
  __builtin_amdgcn_tensor_load_to_lds(g0, g1, z4, z4, z8, 0);
#else
  __builtin_amdgcn_tensor_load_to_lds(g0, g1, z4, z4, 0);
#endif
  __builtin_amdgcn_s_wait_tensorcnt(0);
}
#endif

// ---------------------------------------------------------------------------
// Generic block-level GEMM, N == 64.  C[m,n] = act(A[m,:] @ Wt[n,:]^T + b[n])
// A supplied by a gather lambda (f32 -> f16), Wt is prepacked f16 [N][K].
// One wave per 16x16 tile, waves strided over tiles.
// ---------------------------------------------------------------------------
template <typename FA, typename FD>
__device__ void gemm_block64(FA loadA, const _Float16* __restrict__ Wt,
                             const float* __restrict__ bias, FD storeD,
                             int M, int K) {
  const int wave = threadIdx.x >> 5;
  const int lane = threadIdx.x & 31;
  const int hl = lane & 15, hi = lane >> 4;
  const int tiles = (M >> 4) * 4;
  const int nw = blockDim.x >> 5;
  for (int tile = wave; tile < tiles; tile += nw) {
    const int mt = tile >> 2, nt = tile & 3;
    const int row = mt * 16 + hl;
    const int col = nt * 16 + hl;
    v8f acc = {0.f, 0.f, 0.f, 0.f, 0.f, 0.f, 0.f, 0.f};
    for (int kc = 0; kc < K; kc += 32) {
      Frag a, b;
#pragma unroll
      for (int q = 0; q < 8; ++q) {
        const int k = kc + kqoff(q, hi);
        a.u[q] = pack2(loadA(row, k), loadA(row, k + 1));
        b.u[q] = *(const unsigned*)(Wt + col * K + k);
      }
      acc = wmma_f16(a, b, acc);
    }
    const float bn = bias[col];
#pragma unroll
    for (int e = 0; e < 8; ++e)
      storeD(mt * 16 + hi * 8 + e, col, acc[e] + bn);
  }
}

// ---------------------------------------------------------------------------
// Weight packing kernels (run once per call; cheap).
// ---------------------------------------------------------------------------
__global__ void pack_lin(const float* __restrict__ W, _Float16* __restrict__ Wt,
                         int K, int N) {
  int i = blockIdx.x * 256 + threadIdx.x;
  if (i >= K * N) return;
  int n = i / K, k = i % K;
  Wt[i] = (_Float16)W[k * N + n];             // Wt[n][k] = W[k][n]
}

// conv0: K-layout k = r*16 + ci, padded to 160 (r<9, ci<10 real; rest zero).
__global__ void pack_conv0(const float* __restrict__ W, _Float16* __restrict__ Wg) {
  int i = blockIdx.x * 256 + threadIdx.x;
  if (i >= 128 * 160) return;
  int co = i / 160, k = i % 160;
  int r = k >> 4, ci = k & 15;
  Wg[i] = (r < 9 && ci < 10) ? (_Float16)W[co * 90 + ci * 9 + r] : (_Float16)0.0f;
}

// conv1-4: K-layout k = r*128 + ci (exact, K = 1152).
__global__ void pack_convN(const float* __restrict__ W, _Float16* __restrict__ Wg) {
  int i = blockIdx.x * 256 + threadIdx.x;
  if (i >= 128 * 1152) return;
  int co = i / 1152, k = i % 1152;
  int r = k >> 7, ci = k & 127;
  Wg[i] = (_Float16)W[(co * 128 + ci) * 9 + r];
}

// ---------------------------------------------------------------------------
// Conv 3x3 SAME (+bias, +optional residual, ReLU) fused with 2x2 maxpool.
// Implicit GEMM: M = 128 out channels (8 waves x 16), N = 16 positions
// (2 rows x 8 cols of the pre-pool grid), K staged in KT-wide tiles via TDM.
// Index clamping is safe: padded K positions carry zero weights.
// ---------------------------------------------------------------------------
__device__ __forceinline__ float conv_in(
    const float* __restrict__ x, const float* __restrict__ xc,
    const float* __restrict__ yc, const float* __restrict__ hin,
    int first, int s, int ci, int iy, int ix, int H, int W, int Cin) {
  if (iy < 0 || iy >= H || ix < 0 || ix >= W) return 0.f;
  if (!first) return hin[((s * Cin + ci) * H + iy) * W + ix];
  // first layer: channels 0-3 frame i, 4-7 frame i+1, 8 x_cor, 9 y_cor
  const int i = s >> 6, b = s & 63;
  if (ci < 4)  return x[(((((i    ) * 64 + b) * 4 + ci      )) << 10) + (iy << 5) + ix];
  if (ci < 8)  return x[(((((i + 1) * 64 + b) * 4 + (ci - 4))) << 10) + (iy << 5) + ix];
  if (ci == 8) return xc[(s << 10) + (iy << 5) + ix];
  return yc[(s << 10) + (iy << 5) + ix];
}

__global__ __launch_bounds__(256)
void conv_pool_kernel(const float* __restrict__ x, const float* __restrict__ xc,
                      const float* __restrict__ yc, const float* __restrict__ hin,
                      const _Float16* __restrict__ Wg, const float* __restrict__ bias,
                      float* __restrict__ hout,
                      int H, int W, int Cin, int Kpad, int KT,
                      int cishift, int cimask, int first) {
  __shared__ __align__(16) _Float16 aLds[128 * 160];  // weights tile [co][kl]
  __shared__ __align__(16) _Float16 bLds[16 * 160];   // im2col tile  [n][kl]
  __shared__ float oLds[16 * 128];                    // pre-pool acts [n][co]
  const int sample = blockIdx.y;
  const int tilesX = (W >= 8) ? (W >> 3) : 1;
  const int tr = blockIdx.x / tilesX, tc = blockIdx.x % tilesX;
  const int wave = threadIdx.x >> 5, lane = threadIdx.x & 31;
  const int hl = lane & 15, hi = lane >> 4;
#ifdef HAVE_TDM
  const unsigned aOff = (unsigned)(size_t)(&aLds[0]);
#endif

  v8f acc = {0.f, 0.f, 0.f, 0.f, 0.f, 0.f, 0.f, 0.f};
  for (int kt = 0; kt < Kpad; kt += KT) {
#ifdef HAVE_TDM
    if (wave == 0)  // one TDM issue per workgroup; barrier publishes the tile
      tdm_load_a_tile(Wg + kt, aOff, KT, 128, Kpad);
#else
    for (int idx = threadIdx.x; idx < 128 * (KT >> 3); idx += 256) {
      const int rrow = idx / (KT >> 3);
      const int cc = (idx - rrow * (KT >> 3)) << 3;
      *(uint4*)(aLds + rrow * KT + cc) =
          *(const uint4*)(Wg + rrow * Kpad + kt + cc);
    }
#endif
    // stage B: [16 positions][KT halves], 2 halves per thread per pass
    for (int idx = threadIdx.x; idx < (KT << 3); idx += 256) {
      const int n = idx & 15;
      const int kl0 = (idx >> 4) << 1;
      const int yy = 2 * tr + (n >> 3), xx = 8 * tc + (n & 7);
      float f[2];
#pragma unroll
      for (int j = 0; j < 2; ++j) {
        const int k = kt + kl0 + j;
        int r = k >> cishift;  if (r > 8) r = 8;          // padded -> zero weight
        int ci = k & cimask;   if (ci >= Cin) ci = Cin - 1;
        const int ky = r / 3, kx = r % 3;
        f[j] = conv_in(x, xc, yc, hin, first, sample, ci,
                       yy + ky - 1, xx + kx - 1, H, W, Cin);
      }
      *(unsigned*)(bLds + n * KT + kl0) = pack2(f[0], f[1]);
    }
    __syncthreads();
    for (int kc = 0; kc < KT; kc += 32) {
      Frag a, b;
#pragma unroll
      for (int q = 0; q < 8; ++q) {
        const int k = kc + kqoff(q, hi);
        a.u[q] = *(const unsigned*)(aLds + (wave * 16 + hl) * KT + k);
        b.u[q] = *(const unsigned*)(bLds + hl * KT + k);
      }
      acc = wmma_f16(a, b, acc);
    }
    __syncthreads();
  }
  {  // epilogue: bias (+residual) + relu into oLds
    const int n = hl;
    const int yy = 2 * tr + (n >> 3), xx = 8 * tc + (n & 7);
    const bool valid = (xx < W);
#pragma unroll
    for (int e = 0; e < 8; ++e) {
      const int co = wave * 16 + hi * 8 + e;
      float v = acc[e] + bias[co];
      if (valid) {
        if (!first) v += hin[((sample * Cin + co) * H + yy) * W + xx];
        oLds[(n << 7) + co] = fmaxf(v, 0.f);
      }
    }
  }
  __syncthreads();
  {  // fused 2x2 maxpool -> hout [320,128,H/2,W/2]
    const int Ho = H >> 1, Wo = W >> 1;
    for (int idx = threadIdx.x; idx < 512; idx += 256) {
      const int co = idx & 127, p = idx >> 7;
      const int ox = tc * 4 + p, oy = tr;
      if (ox < Wo) {
        float m0 = oLds[((2 * p) << 7) + co];
        float m1 = oLds[((2 * p + 1) << 7) + co];
        float m2 = oLds[((8 + 2 * p) << 7) + co];
        float m3 = oLds[((8 + 2 * p + 1) << 7) + co];
        hout[((sample * 128 + co) * Ho + oy) * Wo + ox] =
            fmaxf(fmaxf(m0, m1), fmaxf(m2, m3));
      }
    }
  }
}

// ---------------------------------------------------------------------------
// Generic global-memory linear layer (one wave per 16x16 tile).
// out = act( maybe_relu(A) @ W + b [+ maybe_relu(res)] )
// ---------------------------------------------------------------------------
__global__ __launch_bounds__(256)
void lin_kernel(const float* __restrict__ A, int reluA,
                const _Float16* __restrict__ Wt, const float* __restrict__ bias,
                const float* __restrict__ res, int reluRes,
                float* __restrict__ out, int reluOut,
                int M, int K, int N) {
  const int wave = threadIdx.x >> 5, lane = threadIdx.x & 31;
  const int hl = lane & 15, hi = lane >> 4;
  const int nT = N >> 4;
  const int tiles = (M >> 4) * nT;
  const int tile = blockIdx.x * 8 + wave;
  if (tile >= tiles) return;
  const int mt = tile / nT, nt = tile % nT;
  const int row = mt * 16 + hl, col = nt * 16 + hl;
  v8f acc = {0.f, 0.f, 0.f, 0.f, 0.f, 0.f, 0.f, 0.f};
  for (int kc = 0; kc < K; kc += 32) {
    Frag a, b;
#pragma unroll
    for (int q = 0; q < 8; ++q) {
      const int k = kc + kqoff(q, hi);
      float a0 = A[row * K + k], a1 = A[row * K + k + 1];
      if (reluA) { a0 = fmaxf(a0, 0.f); a1 = fmaxf(a1, 0.f); }
      a.u[q] = pack2(a0, a1);
      b.u[q] = *(const unsigned*)(Wt + col * K + k);
    }
    acc = wmma_f16(a, b, acc);
  }
  const float bn = bias[col];
#pragma unroll
  for (int e = 0; e < 8; ++e) {
    const int m = mt * 16 + hi * 8 + e;
    float v = acc[e] + bn;
    if (res) { float r = res[m * N + col]; if (reluRes) r = fmaxf(r, 0.f); v += r; }
    if (reluOut) v = fmaxf(v, 0.f);
    out[m * N + col] = v;
  }
}

// ---------------------------------------------------------------------------
// diff[(t*64+b)*3+j][0:64]  = enc[t,b,j,:], [64:128] = enc[t+1,b,j,:]
// ---------------------------------------------------------------------------
__global__ void build_diff(const float* __restrict__ enc, float* __restrict__ diff) {
  int idx = blockIdx.x * 256 + threadIdx.x;
  if (idx >= 768 * 128) return;
  int row = idx >> 7, k = idx & 127;
  int t = row / 192, rem = row % 192;
  int b = rem / 3, j = rem % 3;
  float v = (k < 64) ? enc[((t)     * 64 + b) * 192 + j * 64 + k]
                     : enc[((t + 1) * 64 + b) * 192 + j * 64 + (k - 64)];
  diff[idx] = v;
}

__global__ void add_kernel(const float* __restrict__ a, const float* __restrict__ b,
                           float* __restrict__ o, int n) {
  int i = blockIdx.x * 256 + threadIdx.x;
  if (i < n) o[i] = a[i] + b[i];
}

// ---------------------------------------------------------------------------
// Relational core: one workgroup per (core z, batch chunk of 16).
// All intermediates LDS-resident: rA/rB 96x64, rC/rD 48x64 (72 KB).
// ---------------------------------------------------------------------------
struct CoreP  { const _Float16* w[10]; const float* b[10]; };
struct CoreArgs { CoreP z[3]; };

__global__ __launch_bounds__(512)
void core_kernel(CoreArgs cp, const float* __restrict__ stIn,
                 float* __restrict__ cbuf) {
  extern __shared__ float lds[];
  float* rA = lds;                    // 96*64
  float* rB = rA + 96 * 64;           // 96*64
  float* rC = rB + 96 * 64;           // 48*64
  float* rD = rC + 48 * 64;           // 48*64
  const CoreP& p = cp.z[blockIdx.x];
  const int bc = blockIdx.y;          // batch chunk (16 of 64)
  const float* s = stIn + blockIdx.x * (192 * 64) + bc * (48 * 64);
  float* cz = cbuf + blockIdx.x * (192 * 64) + bc * (48 * 64);

  // h1 = relu(s @ sc0)                                  -> rD
  gemm_block64([=](int r, int k) { return s[(r << 6) + k]; }, p.w[0], p.b[0],
               [=](int m, int n, float v) { rD[(m << 6) + n] = fmaxf(v, 0.f); },
               48, 64);
  __syncthreads();
  // self_dyn = (h1 @ sc1) + h1                          -> rC
  gemm_block64([=](int r, int k) { return rD[(r << 6) + k]; }, p.w[1], p.b[1],
               [=](int m, int n, float v) { rC[(m << 6) + n] = v + rD[(m << 6) + n]; },
               48, 64);
  __syncthreads();
  // r1 = relu(concat(s[ROWS], s[COLS]) @ rc0)           -> rA  (M=96, K=128)
  gemm_block64([=](int r, int k) {
                 const int b = r / 6, j = r % 6;     // ROWS[j]=j/2, COLS[j]=j%2
                 return (k < 64) ? s[((b * 3 + (j >> 1)) << 6) + k]
                                 : s[((b * 3 + (j & 1)) << 6) + (k - 64)];
               }, p.w[2], p.b[2],
               [=](int m, int n, float v) { rA[(m << 6) + n] = fmaxf(v, 0.f); },
               96, 128);
  __syncthreads();
  // r2 = relu((r1 @ rc1) + r1)                          -> rB
  gemm_block64([=](int r, int k) { return rA[(r << 6) + k]; }, p.w[3], p.b[3],
               [=](int m, int n, float v) {
                 rB[(m << 6) + n] = fmaxf(v + rA[(m << 6) + n], 0.f); },
               96, 64);
  __syncthreads();
  // r3 = (r2 @ rc2) + r2                                -> rA
  gemm_block64([=](int r, int k) { return rB[(r << 6) + k]; }, p.w[4], p.b[4],
               [=](int m, int n, float v) { rA[(m << 6) + n] = v + rB[(m << 6) + n]; },
               96, 64);
  __syncthreads();
  // dyn = self_dyn + r3[2i] + r3[2i+1]                  -> rD
  for (int idx = threadIdx.x; idx < 48 * 64; idx += blockDim.x) {
    const int row = idx >> 6, n = idx & 63;
    const int b = row / 3, i = row % 3;
    rD[idx] = rC[idx] + rA[((b * 6 + 2 * i) << 6) + n]
                      + rA[((b * 6 + 2 * i + 1) << 6) + n];
  }
  __syncthreads();
  // a1 = relu(dyn @ af0)                                -> rB
  gemm_block64([=](int r, int k) { return rD[(r << 6) + k]; }, p.w[5], p.b[5],
               [=](int m, int n, float v) { rB[(m << 6) + n] = fmaxf(v, 0.f); },
               48, 64);
  __syncthreads();
  // a2 = relu((a1 @ af1) + a1)                          -> rA
  gemm_block64([=](int r, int k) { return rB[(r << 6) + k]; }, p.w[6], p.b[6],
               [=](int m, int n, float v) {
                 rA[(m << 6) + n] = fmaxf(v + rB[(m << 6) + n], 0.f); },
               48, 64);
  __syncthreads();
  // a3 = (a2 @ af2) + dyn                               -> rB
  gemm_block64([=](int r, int k) { return rA[(r << 6) + k]; }, p.w[7], p.b[7],
               [=](int m, int n, float v) { rB[(m << 6) + n] = v + rD[(m << 6) + n]; },
               48, 64);
  __syncthreads();
  // o1 = relu(concat(a3, s) @ oc0)                      -> rA  (K=128)
  gemm_block64([=](int r, int k) {
                 return (k < 64) ? rB[(r << 6) + k] : s[(r << 6) + (k - 64)];
               }, p.w[8], p.b[8],
               [=](int m, int n, float v) { rA[(m << 6) + n] = fmaxf(v, 0.f); },
               48, 128);
  __syncthreads();
  // c = (o1 @ oc1) + o1                                 -> global cbuf
  gemm_block64([=](int r, int k) { return rA[(r << 6) + k]; }, p.w[9], p.b[9],
               [=](int m, int n, float v) { cz[(m << 6) + n] = v + rA[(m << 6) + n]; },
               48, 64);
}

// ---------------------------------------------------------------------------
// Aggregator: agg = lin(relu(lin(concat(c1,c2,c3), agg1)), agg2)
// Writes both the next scan state and rolls[t] (d_out).
// ---------------------------------------------------------------------------
__global__ __launch_bounds__(512)
void agg_kernel(const float* __restrict__ cb,
                const _Float16* __restrict__ w1, const float* __restrict__ b1,
                const _Float16* __restrict__ w2, const float* __restrict__ b2,
                float* __restrict__ stOut, float* __restrict__ roll) {
  __shared__ float t1[192 * 64];
  const float* c1 = cb;
  const float* c2 = cb + 192 * 64;
  const float* c3 = cb + 2 * 192 * 64;
  gemm_block64([=](int r, int k) {
                 return (k < 64) ? c1[(r << 6) + k]
                      : (k < 128) ? c2[(r << 6) + (k - 64)]
                                  : c3[(r << 6) + (k - 128)];
               }, w1, b1,
               [&](int m, int n, float v) { t1[(m << 6) + n] = fmaxf(v, 0.f); },
               192, 192);
  __syncthreads();
  gemm_block64([&](int r, int k) { return t1[(r << 6) + k]; }, w2, b2,
               [=](int m, int n, float v) {
                 stOut[(m << 6) + n] = v;
                 roll[(m << 6) + n] = v; },
               192, 64);
}

// ---------------------------------------------------------------------------
// Host orchestration
// ---------------------------------------------------------------------------
extern "C" void kernel_launch(void* const* d_in, const int* in_sizes, int n_in,
                              void* d_out, int out_size, void* d_ws, size_t ws_size,
                              hipStream_t stream) {
  (void)out_size;
  if (n_in < 85) return;
  const float* x    = (const float*)d_in[0];
  const float* xcor = (const float*)d_in[1];
  const float* ycor = (const float*)d_in[2];
  auto P = [&](int i) { return (const float*)d_in[i]; };

  // Param index map: harness may flatten the params dict in insertion order
  // or sorted-key order; in_sizes[3] disambiguates (11520 = conv0.W, 4096 = af.W).
  const bool srt = (in_sizes[3] == 4096);
  int iConvW[5], iConvB[5], iFc[3][2];
  int iSc[3][2][2], iRc[3][3][2], iAf[3][3][2], iOc[3][2][2], iAg1[2], iAg2[2];
  if (!srt) {
    for (int i = 0; i < 5; ++i) { iConvW[i] = 3 + 2 * i; iConvB[i] = 4 + 2 * i; }
    for (int f = 0; f < 3; ++f) { iFc[f][0] = 13 + 2 * f; iFc[f][1] = 14 + 2 * f; }
    for (int i = 0; i < 3; ++i) {
      for (int j = 0; j < 2; ++j) { iSc[i][j][0] = 21 + i * 4 + j * 2; iSc[i][j][1] = iSc[i][j][0] + 1; }
      for (int j = 0; j < 3; ++j) { iRc[i][j][0] = 33 + i * 6 + j * 2; iRc[i][j][1] = iRc[i][j][0] + 1; }
      for (int j = 0; j < 3; ++j) { iAf[i][j][0] = 51 + i * 6 + j * 2; iAf[i][j][1] = iAf[i][j][0] + 1; }
      for (int j = 0; j < 2; ++j) { iOc[i][j][0] = 69 + i * 4 + j * 2; iOc[i][j][1] = iOc[i][j][0] + 1; }
    }
    iAg1[0] = 81; iAg1[1] = 82; iAg2[0] = 83; iAg2[1] = 84;
  } else {
    for (int i = 0; i < 3; ++i) {
      for (int j = 0; j < 3; ++j) { iAf[i][j][0] = 3 + i * 6 + j * 2;  iAf[i][j][1] = iAf[i][j][0] + 1; }
      for (int j = 0; j < 2; ++j) { iOc[i][j][0] = 43 + i * 4 + j * 2; iOc[i][j][1] = iOc[i][j][0] + 1; }
      for (int j = 0; j < 3; ++j) { iRc[i][j][0] = 55 + i * 6 + j * 2; iRc[i][j][1] = iRc[i][j][0] + 1; }
      for (int j = 0; j < 2; ++j) { iSc[i][j][0] = 73 + i * 4 + j * 2; iSc[i][j][1] = iSc[i][j][0] + 1; }
    }
    iAg1[0] = 21; iAg1[1] = 22; iAg2[0] = 23; iAg2[1] = 24;
    for (int i = 0; i < 5; ++i) { iConvW[i] = 25 + 2 * i; iConvB[i] = 26 + 2 * i; }
    for (int f = 0; f < 3; ++f) { iFc[f][0] = 35 + 2 * f; iFc[f][1] = 36 + 2 * f; }
  }

  // ----- scratch carve-out -----
  char* ws = (char*)d_ws;
  size_t off = 0;
  auto alloc = [&](size_t bytes) -> void* {
    off = (off + 255) & ~(size_t)255;
    void* p = ws + off;
    off += bytes;
    return p;
  };
  _Float16* Wgc[5];
  Wgc[0] = (_Float16*)alloc(128 * 160 * 2);
  for (int i = 1; i < 5; ++i) Wgc[i] = (_Float16*)alloc(128 * 1152 * 2);
  _Float16* Tfc1 = (_Float16*)alloc(192 * 128 * 2);
  _Float16* Tfc2 = (_Float16*)alloc(64 * 128 * 2);
  _Float16* Tfc3 = (_Float16*)alloc(64 * 64 * 2);
  _Float16 *Tsc[3][2], *Trc[3][3], *Taf[3][3], *Toc[3][2];
  for (int z = 0; z < 3; ++z) {
    for (int j = 0; j < 2; ++j) Tsc[z][j] = (_Float16*)alloc(64 * 64 * 2);
    Trc[z][0] = (_Float16*)alloc(64 * 128 * 2);
    Trc[z][1] = (_Float16*)alloc(64 * 64 * 2);
    Trc[z][2] = (_Float16*)alloc(64 * 64 * 2);
    for (int j = 0; j < 3; ++j) Taf[z][j] = (_Float16*)alloc(64 * 64 * 2);
    Toc[z][0] = (_Float16*)alloc(64 * 128 * 2);
    Toc[z][1] = (_Float16*)alloc(64 * 64 * 2);
  }
  _Float16* Tag1 = (_Float16*)alloc(64 * 192 * 2);
  _Float16* Tag2 = (_Float16*)alloc(64 * 64 * 2);
  float* h0   = (float*)alloc((size_t)320 * 128 * 16 * 16 * 4);
  float* h1   = (float*)alloc((size_t)320 * 128 * 8 * 8 * 4);
  float* h2   = (float*)alloc((size_t)320 * 128 * 4 * 4 * 4);
  float* h3   = (float*)alloc((size_t)320 * 128 * 2 * 2 * 4);
  float* unit = (float*)alloc((size_t)320 * 128 * 4);
  float* enc  = (float*)alloc((size_t)320 * 192 * 4);
  float* diff = (float*)alloc((size_t)768 * 128 * 4);
  float* g1   = (float*)alloc((size_t)768 * 64 * 4);
  float* sh2  = (float*)alloc((size_t)768 * 64 * 4);
  float* st   = (float*)alloc((size_t)24 * 192 * 64 * 4);   // states sh3[0..3] + 20 aggs
  float* cb   = (float*)alloc((size_t)3 * 192 * 64 * 4);    // c1,c2,c3
  if (off > ws_size) return;   // insufficient scratch: bail deterministically

  // ----- weight packing -----
  auto packL = [&](const float* W, _Float16* dst, int K, int N) {
    int n = K * N;
    pack_lin<<<dim3((n + 255) / 256), dim3(256), 0, stream>>>(W, dst, K, N);
  };
  pack_conv0<<<dim3((128 * 160 + 255) / 256), dim3(256), 0, stream>>>(
      P(iConvW[0]), Wgc[0]);
  for (int i = 1; i < 5; ++i)
    pack_convN<<<dim3((128 * 1152 + 255) / 256), dim3(256), 0, stream>>>(
        P(iConvW[i]), Wgc[i]);
  packL(P(iFc[0][0]), Tfc1, 128, 192);
  packL(P(iFc[1][0]), Tfc2, 128, 64);
  packL(P(iFc[2][0]), Tfc3, 64, 64);
  for (int z = 0; z < 3; ++z) {
    packL(P(iSc[z][0][0]), Tsc[z][0], 64, 64);
    packL(P(iSc[z][1][0]), Tsc[z][1], 64, 64);
    packL(P(iRc[z][0][0]), Trc[z][0], 128, 64);
    packL(P(iRc[z][1][0]), Trc[z][1], 64, 64);
    packL(P(iRc[z][2][0]), Trc[z][2], 64, 64);
    for (int j = 0; j < 3; ++j) packL(P(iAf[z][j][0]), Taf[z][j], 64, 64);
    packL(P(iOc[z][0][0]), Toc[z][0], 128, 64);
    packL(P(iOc[z][1][0]), Toc[z][1], 64, 64);
  }
  packL(P(iAg1[0]), Tag1, 192, 64);
  packL(P(iAg2[0]), Tag2, 64, 64);

  // ----- conv stack (implicit GEMM + TDM weight staging + fused pool) -----
  auto convL = [&](const float* hin, _Float16* Wg, const float* bias, float* hout,
                   int H, int W, int Cin, int Kpad, int KT, int cish, int cimask,
                   int first) {
    int tilesX = (W >= 8) ? (W / 8) : 1;
    int tiles = (H / 2) * tilesX;
    conv_pool_kernel<<<dim3(tiles, 320), dim3(256), 0, stream>>>(
        x, xcor, ycor, hin, Wg, bias, hout, H, W, Cin, Kpad, KT, cish, cimask,
        first);
  };
  convL(nullptr, Wgc[0], P(iConvB[0]), h0,   32, 32,  10,  160, 160, 4,  15, 1);
  convL(h0,      Wgc[1], P(iConvB[1]), h1,   16, 16, 128, 1152, 128, 7, 127, 0);
  convL(h1,      Wgc[2], P(iConvB[2]), h2,    8,  8, 128, 1152, 128, 7, 127, 0);
  convL(h2,      Wgc[3], P(iConvB[3]), h3,    4,  4, 128, 1152, 128, 7, 127, 0);
  convL(h3,      Wgc[4], P(iConvB[4]), unit,  2,  2, 128, 1152, 128, 7, 127, 0);

  // ----- encoder MLP -----
  auto linL = [&](const float* A, int rA_, const _Float16* Wt, const float* b,
                  const float* res, int rR, float* o, int rO, int M, int K, int N) {
    int tiles = (M / 16) * (N / 16);
    lin_kernel<<<dim3((tiles + 7) / 8), dim3(256), 0, stream>>>(
        A, rA_, Wt, b, res, rR, o, rO, M, K, N);
  };
  linL(unit, 0, Tfc1, P(iFc[0][1]), nullptr, 0, enc, 0, 320, 128, 192);
  build_diff<<<dim3((768 * 128 + 255) / 256), dim3(256), 0, stream>>>(enc, diff);
  // g1 = lin(diff, fc2)   (pre-relu; reused for sh1 and sh3)
  linL(diff, 0, Tfc2, P(iFc[1][1]), nullptr, 0, g1, 0, 768, 128, 64);
  // sh2 = relu(lin(relu(g1), fc3) + relu(g1))
  linL(g1, 1, Tfc3, P(iFc[2][1]), g1, 1, sh2, 1, 768, 64, 64);
  // sh3 = g1 + sh2  -> st[0..3]
  add_kernel<<<dim3((768 * 64 + 255) / 256), dim3(256), 0, stream>>>(
      g1, sh2, st, 768 * 64);

  // ----- 20-step relational scan -----
  CoreArgs ca;
  for (int z = 0; z < 3; ++z) {
    ca.z[z].w[0] = Tsc[z][0]; ca.z[z].b[0] = P(iSc[z][0][1]);
    ca.z[z].w[1] = Tsc[z][1]; ca.z[z].b[1] = P(iSc[z][1][1]);
    ca.z[z].w[2] = Trc[z][0]; ca.z[z].b[2] = P(iRc[z][0][1]);
    ca.z[z].w[3] = Trc[z][1]; ca.z[z].b[3] = P(iRc[z][1][1]);
    ca.z[z].w[4] = Trc[z][2]; ca.z[z].b[4] = P(iRc[z][2][1]);
    ca.z[z].w[5] = Taf[z][0]; ca.z[z].b[5] = P(iAf[z][0][1]);
    ca.z[z].w[6] = Taf[z][1]; ca.z[z].b[6] = P(iAf[z][1][1]);
    ca.z[z].w[7] = Taf[z][2]; ca.z[z].b[7] = P(iAf[z][2][1]);
    ca.z[z].w[8] = Toc[z][0]; ca.z[z].b[8] = P(iOc[z][0][1]);
    ca.z[z].w[9] = Toc[z][1]; ca.z[z].b[9] = P(iOc[z][1][1]);
  }
  const size_t coreLds = (2 * 96 * 64 + 2 * 48 * 64) * sizeof(float);  // 72 KB
  float* outF = (float*)d_out;
  for (int t = 0; t < 20; ++t) {
    core_kernel<<<dim3(3, 4), dim3(512), coreLds, stream>>>(
        ca, st + (size_t)t * 192 * 64, cb);
    agg_kernel<<<dim3(1), dim3(512), 0, stream>>>(
        cb, Tag1, P(iAg1[1]), Tag2, P(iAg2[1]),
        st + (size_t)(t + 4) * 192 * 64, outF + (size_t)t * 192 * 64);
  }
}
